// Seq2seq_30425548324886
// MI455X (gfx1250) — compile-verified, matrix operand order checked by reading, not run on hardware
//
#include <hip/hip_runtime.h>
#include <hip/hip_bf16.h>

// ---------------------------------------------------------------------------
// Types for CDNA5 WMMA (wave32, v_wmma_f32_16x16x32_bf16: A 16x32, B 32x16,
// C/D 16x16 f32 in 8 VGPRs).
// ---------------------------------------------------------------------------
typedef __bf16 bf16_t;
typedef __attribute__((ext_vector_type(16))) __bf16 v16bf;
typedef __attribute__((ext_vector_type(8)))  float  v8f;
typedef __attribute__((ext_vector_type(4)))  float  f32x4;

// Per-lane fragment load for 16-bit A (16x32) / B^T (16 rows x 32 K) tiles.
// ISA layout (CDNA5 ISA): lanes 0-15 => rows 0-15, K = {0..7,16..23};
// lanes 16-31 => rows 0-15, K = {8..15,24..31}. Two 16-byte loads per lane.
static __device__ __forceinline__ v16bf load_frag32(const bf16_t* __restrict__ row,
                                                    int kc, int half) {
  union { f32x4 q[2]; v16bf v; } u;
  u.q[0] = *reinterpret_cast<const f32x4*>(row + kc + half * 8);
  u.q[1] = *reinterpret_cast<const f32x4*>(row + kc + half * 8 + 16);
  return u.v;
}

static __device__ __forceinline__ v8f wmma_bf16(v16bf a, v16bf b, v8f c) {
  return __builtin_amdgcn_wmma_f32_16x16x32_bf16(
      /*neg_a=*/false, a, /*neg_b=*/false, b,
      /*c_mod=*/(short)0, c, /*reuse_a=*/false, /*reuse_b=*/false);
}

// ---------------------------------------------------------------------------
// Generic bf16 GEMM:  C[M x N] = A[M x K] * W[N x K]^T (+ bias[N]) (+ base)
// Block = 128 threads = 4 waves. Each wave computes a 16x64 strip (4 WMMA
// accumulators, A fragment reused across the 4 N-subtiles).
// Double-buffered k-loop: all 5 fragments of step k+1 are issued before the
// 4 back-to-back WMMAs of step k, so s_wait_loadcnt covers loads issued a
// full iteration earlier. Unconditional speculative global_prefetch on the
// streamed W rows. Grid: (N/256, M/16). K % 32 == 0.
// ---------------------------------------------------------------------------
__global__ __launch_bounds__(128) void wmma_gemm_nt(
    const bf16_t* __restrict__ A, int lda,
    const bf16_t* __restrict__ W, int K,
    const float* __restrict__ base, long long base_stride,
    const float* __restrict__ bias,
    float* __restrict__ C, long long ldc)
{
  const int lane = threadIdx.x & 31;
  const int wave = threadIdx.x >> 5;
  const int r    = lane & 15;     // row/col within 16
  const int hh   = lane >> 4;     // lane half selects K sub-pattern
  const int m0   = blockIdx.y << 4;
  const int n0   = (blockIdx.x * 4 + wave) << 6;

  const v8f vzero = {0.f, 0.f, 0.f, 0.f, 0.f, 0.f, 0.f, 0.f};
  v8f acc0 = vzero, acc1 = vzero, acc2 = vzero, acc3 = vzero;

  const bf16_t* Arow  = A + (size_t)(m0 + r) * lda;
  const bf16_t* Wrow0 = W + (size_t)(n0 + r) * K;
  const bf16_t* Wrow1 = Wrow0 + (size_t)16 * K;
  const bf16_t* Wrow2 = Wrow0 + (size_t)32 * K;
  const bf16_t* Wrow3 = Wrow0 + (size_t)48 * K;

  // prologue: fragments for k-step 0
  v16bf a  = load_frag32(Arow,  0, hh);
  v16bf b0 = load_frag32(Wrow0, 0, hh);
  v16bf b1 = load_frag32(Wrow1, 0, hh);
  v16bf b2 = load_frag32(Wrow2, 0, hh);
  v16bf b3 = load_frag32(Wrow3, 0, hh);

#pragma unroll 2
  for (int kc = 0; kc < K; kc += 32) {
    // speculative prefetch ~8 k-steps ahead on the streamed W rows
    __builtin_prefetch(Wrow0 + kc + 256, 0, 1);
    __builtin_prefetch(Wrow1 + kc + 256, 0, 1);
    __builtin_prefetch(Wrow2 + kc + 256, 0, 1);
    __builtin_prefetch(Wrow3 + kc + 256, 0, 1);

    // branchless next-step index (last iteration harmlessly reloads k=0)
    const int kn = (kc + 32 < K) ? (kc + 32) : 0;
    v16bf an  = load_frag32(Arow,  kn, hh);
    v16bf bn0 = load_frag32(Wrow0, kn, hh);
    v16bf bn1 = load_frag32(Wrow1, kn, hh);
    v16bf bn2 = load_frag32(Wrow2, kn, hh);
    v16bf bn3 = load_frag32(Wrow3, kn, hh);

    // 4 back-to-back WMMAs on the previously loaded fragments
    acc0 = wmma_bf16(a, b0, acc0);
    acc1 = wmma_bf16(a, b1, acc1);
    acc2 = wmma_bf16(a, b2, acc2);
    acc3 = wmma_bf16(a, b3, acc3);

    a = an; b0 = bn0; b1 = bn1; b2 = bn2; b3 = bn3;
  }

  // C/D layout: VGPR v, lanes 0-15 -> M=m0+v, N=n0+16j+r; lanes 16-31 -> M=m0+v+8.
  v8f accv[4] = {acc0, acc1, acc2, acc3};
#pragma unroll
  for (int j = 0; j < 4; ++j) {
    const int col = n0 + (j << 4) + r;
    const float bv = bias ? bias[col] : 0.0f;
#pragma unroll
    for (int v = 0; v < 8; ++v) {
      const int row = m0 + v + hh * 8;
      float val = accv[j][v] + bv;
      if (base) val += base[(size_t)row * base_stride + col];
      C[(size_t)row * ldc + col] = val;
    }
  }
}

// ---------------------------------------------------------------------------
// Elementwise LSTM cell update with mask. Also produces the bf16 state copy
// consumed by the next step's WMMA GEMM, and the masked bf16 sequence output.
// gates: B x 4H (i | f | g | o).  grid*block == B*H exactly.
// ---------------------------------------------------------------------------
__global__ __launch_bounds__(256) void lstm_update(
    const float* __restrict__ gates,
    const unsigned char* __restrict__ mask, int mask_stride, int t,
    float* __restrict__ hbuf, float* __restrict__ cbuf,
    bf16_t* __restrict__ hbf,
    int ldstate, int col_off, int H,
    bf16_t* __restrict__ outbf, long long out_row_stride,
    long long out_t_stride, int out_col_off)
{
  const int idx = blockIdx.x * blockDim.x + threadIdx.x;
  const int b = idx / H;
  const int j = idx - b * H;

  const float* g = gates + (size_t)b * 4 * H;
  const float gi = g[j];
  const float gf = g[H + j];
  const float gg = g[2 * H + j];
  const float go = g[3 * H + j];

  const size_t sidx = (size_t)b * ldstate + col_off + j;
  const float c_old = cbuf[sidx];
  const float h_old = hbuf[sidx];

  const float i_s = 1.0f / (1.0f + __expf(-gi));
  const float f_s = 1.0f / (1.0f + __expf(-gf));
  const float o_s = 1.0f / (1.0f + __expf(-go));
  const float c2  = f_s * c_old + i_s * tanhf(gg);
  const float h2  = o_s * tanhf(c2);

  const bool m  = mask[(size_t)b * mask_stride + t] != 0;
  const float hn = m ? h2 : h_old;
  const float cn = m ? c2 : c_old;

  hbuf[sidx] = hn;
  cbuf[sidx] = cn;
  hbf[sidx]  = (bf16_t)hn;

  if (outbf) {
    outbf[(size_t)b * out_row_stride + (size_t)t * out_t_stride + out_col_off + j] =
        m ? (bf16_t)hn : (bf16_t)0.0f;
  }
}

// Embedding gather, emits bf16 directly (embeddings are only GEMM inputs).
__global__ __launch_bounds__(256) void embed_bf16(
    const int* __restrict__ idx, const float* __restrict__ emb,
    bf16_t* __restrict__ out, int D, long long n)
{
  const long long i = (long long)blockIdx.x * blockDim.x + threadIdx.x;
  if (i >= n) return;
  const long long row = i / D;
  const long long j   = i - row * D;
  out[i] = (bf16_t)emb[(size_t)idx[row] * D + j];
}

__global__ __launch_bounds__(256) void f32_to_bf16(
    const float* __restrict__ in, bf16_t* __restrict__ out, long long n)
{
  const long long i = (long long)blockIdx.x * blockDim.x + threadIdx.x;
  if (i >= n) return;
  out[i] = (bf16_t)in[i];
}

// ---------------------------------------------------------------------------
// Host-side orchestration
// ---------------------------------------------------------------------------
extern "C" void kernel_launch(void* const* d_in, const int* in_sizes, int n_in,
                              void* d_out, int out_size, void* d_ws, size_t ws_size,
                              hipStream_t stream) {
  (void)in_sizes; (void)n_in; (void)out_size; (void)ws_size;
  constexpr int B = 32, S = 64, T = 65, V = 32000, D = 256, H = 512, DO = 1024;
  constexpr int G4H = 4 * H;    // 2048 (encoder gate width)
  constexpr int G4D = 4 * DO;   // 4096 (decoder gate width)

  const int*           src    = (const int*)d_in[0];
  const unsigned char* smask  = (const unsigned char*)d_in[1];
  const int*           tgt    = (const int*)d_in[2];
  const unsigned char* tmask  = (const unsigned char*)d_in[3];
  const float* emb_src = (const float*)d_in[4];
  const float* emb_tgt = (const float*)d_in[5];
  const float* eW0f = (const float*)d_in[6];
  const float* eU0f = (const float*)d_in[7];
  const float* eb0f = (const float*)d_in[8];
  const float* eW0b = (const float*)d_in[9];
  const float* eU0b = (const float*)d_in[10];
  const float* eb0b = (const float*)d_in[11];
  const float* eW1f = (const float*)d_in[12];
  const float* eU1f = (const float*)d_in[13];
  const float* eb1f = (const float*)d_in[14];
  const float* eW1b = (const float*)d_in[15];
  const float* eU1b = (const float*)d_in[16];
  const float* eb1b = (const float*)d_in[17];
  const float* dW0  = (const float*)d_in[18];
  const float* dU0  = (const float*)d_in[19];
  const float* db0  = (const float*)d_in[20];
  const float* dW1  = (const float*)d_in[21];
  const float* dU1  = (const float*)d_in[22];
  const float* db1  = (const float*)d_in[23];
  const float* wW   = (const float*)d_in[24];
  const float* wb   = (const float*)d_in[25];
  float* out = (float*)d_out;

  // ---- workspace carve-out ----
  char* wsp = (char*)d_ws;
  size_t off = 0;
  auto alloc = [&](size_t bytes) -> void* {
    void* p = wsp + off;
    off = (off + bytes + 255) & ~size_t(255);
    return p;
  };
  auto allocb = [&](size_t elems) { return (bf16_t*)alloc(elems * sizeof(bf16_t)); };
  auto allocf = [&](size_t elems) { return (float*)alloc(elems * sizeof(float)); };

  // bf16 weight copies
  bf16_t* weW0f = allocb((size_t)G4H * D);
  bf16_t* weU0f = allocb((size_t)G4H * H);
  bf16_t* weW0b = allocb((size_t)G4H * D);
  bf16_t* weU0b = allocb((size_t)G4H * H);
  bf16_t* weW1f = allocb((size_t)G4H * (2 * H));
  bf16_t* weU1f = allocb((size_t)G4H * H);
  bf16_t* weW1b = allocb((size_t)G4H * (2 * H));
  bf16_t* weU1b = allocb((size_t)G4H * H);
  bf16_t* wdW0  = allocb((size_t)G4D * D);
  bf16_t* wdU0  = allocb((size_t)G4D * DO);
  bf16_t* wdW1  = allocb((size_t)G4D * DO);
  bf16_t* wdU1  = allocb((size_t)G4D * DO);
  bf16_t* wwW   = allocb((size_t)V * DO);     // 64 MB -> resident in 192 MB L2

  // bf16 activations
  bf16_t* xsrc_bf = allocb((size_t)B * S * D);
  bf16_t* xtgt_bf = allocb((size_t)B * T * D);
  bf16_t* x1_bf   = allocb((size_t)B * S * (2 * H));  // encoder L0 concat out
  bf16_t* dout_bf = allocb((size_t)B * T * DO);       // decoder seq out

  // f32 precomputed input gates
  float* gx_f = allocf((size_t)B * S * G4H);
  float* gx_b = allocf((size_t)B * S * G4H);
  float* gxd  = allocf((size_t)B * T * G4D);

  // states: encoder fwd cols [0,H), bwd cols [H,2H) with row stride DO, so the
  // encoder finals land in place as decoder initial states.
  float*  hA = allocf((size_t)B * DO);
  float*  cA = allocf((size_t)B * DO);
  float*  hB = allocf((size_t)B * DO);
  float*  cB = allocf((size_t)B * DO);
  bf16_t* hAbf = allocb((size_t)B * DO);
  bf16_t* hBbf = allocb((size_t)B * DO);

  float* g_step = allocf((size_t)B * G4D);  // per-step gate scratch (max width)

  // zero states each call (graph replays must be deterministic)
  hipMemsetAsync(hA, 0, (size_t)B * DO * sizeof(float), stream);
  hipMemsetAsync(cA, 0, (size_t)B * DO * sizeof(float), stream);
  hipMemsetAsync(hB, 0, (size_t)B * DO * sizeof(float), stream);
  hipMemsetAsync(cB, 0, (size_t)B * DO * sizeof(float), stream);
  hipMemsetAsync(hAbf, 0, (size_t)B * DO * sizeof(bf16_t), stream);
  hipMemsetAsync(hBbf, 0, (size_t)B * DO * sizeof(bf16_t), stream);

  auto cvt = [&](const float* s, bf16_t* d, size_t n) {
    f32_to_bf16<<<dim3((unsigned)((n + 255) / 256)), 256, 0, stream>>>(s, d, (long long)n);
  };
  cvt(eW0f, weW0f, (size_t)G4H * D);
  cvt(eU0f, weU0f, (size_t)G4H * H);
  cvt(eW0b, weW0b, (size_t)G4H * D);
  cvt(eU0b, weU0b, (size_t)G4H * H);
  cvt(eW1f, weW1f, (size_t)G4H * (2 * H));
  cvt(eU1f, weU1f, (size_t)G4H * H);
  cvt(eW1b, weW1b, (size_t)G4H * (2 * H));
  cvt(eU1b, weU1b, (size_t)G4H * H);
  cvt(dW0,  wdW0,  (size_t)G4D * D);
  cvt(dU0,  wdU0,  (size_t)G4D * DO);
  cvt(dW1,  wdW1,  (size_t)G4D * DO);
  cvt(dU1,  wdU1,  (size_t)G4D * DO);
  cvt(wW,   wwW,   (size_t)V * DO);

  {
    long long n = (long long)B * S * D;
    embed_bf16<<<dim3((unsigned)((n + 255) / 256)), 256, 0, stream>>>(src, emb_src, xsrc_bf, D, n);
    n = (long long)B * T * D;
    embed_bf16<<<dim3((unsigned)((n + 255) / 256)), 256, 0, stream>>>(tgt, emb_tgt, xtgt_bf, D, n);
  }

  auto gemm = [&](const bf16_t* A, int lda, const bf16_t* W, int K,
                  const float* base, long long bstride, const float* bias,
                  float* C, long long ldc, int M, int N) {
    wmma_gemm_nt<<<dim3((unsigned)(N / 256), (unsigned)(M / 16)), 128, 0, stream>>>(
        A, lda, W, K, base, bstride, bias, C, ldc);
  };
  auto update = [&](const float* gates, const unsigned char* mask, int mstride, int t,
                    float* hbuf, float* cbuf, bf16_t* hbf, int col_off, int Hc,
                    bf16_t* obf, long long ors, long long ots, int ocol) {
    lstm_update<<<dim3((unsigned)((B * Hc) / 256)), 256, 0, stream>>>(
        gates, mask, mstride, t, hbuf, cbuf, hbf, DO, col_off, Hc, obf, ors, ots, ocol);
  };

  // ---- precompute decoder layer-0 input gates (independent) ----
  gemm(xtgt_bf, D, wdW0, D, nullptr, 0, db0, gxd, G4D, B * T, G4D);

  // ---- encoder layer 0: gates_x for both directions ----
  gemm(xsrc_bf, D, weW0f, D, nullptr, 0, eb0f, gx_f, G4H, B * S, G4H);
  gemm(xsrc_bf, D, weW0b, D, nullptr, 0, eb0b, gx_b, G4H, B * S, G4H);

  // one encoder-L0 direction: recurrent WMMA GEMM + masked cell update per step
  auto run_enc_dir = [&](const bf16_t* U, const float* gx, int col_off, bool rev) {
    for (int s = 0; s < S; ++s) {
      const int t = rev ? (S - 1 - s) : s;
      gemm(hAbf + col_off, DO, U, H,
           gx + (size_t)t * G4H, (long long)S * G4H, nullptr,
           g_step, G4H, B, G4H);
      update(g_step, smask, S, t, hA, cA, hAbf, col_off, H,
             x1_bf, (long long)S * (2 * H), 2 * H, col_off);
    }
  };

  run_enc_dir(weU0f, gx_f, 0, false);   // L0 forward  -> x1_bf[:, :, 0:512]
  run_enc_dir(weU0b, gx_b, H, true);    // L0 backward -> x1_bf[:, :, 512:1024]

  // ---- encoder layer 1: input gates on concat(of, ob), reuse gx buffers ----
  gemm(x1_bf, 2 * H, weW1f, 2 * H, nullptr, 0, eb1f, gx_f, G4H, B * S, G4H);
  gemm(x1_bf, 2 * H, weW1b, 2 * H, nullptr, 0, eb1b, gx_b, G4H, B * S, G4H);

  for (int s = 0; s < S; ++s) {  // L1 forward (seq output feeds decoder via states)
    gemm(hBbf, DO, weU1f, H, gx_f + (size_t)s * G4H, (long long)S * G4H, nullptr,
         g_step, G4H, B, G4H);
    update(g_step, smask, S, s, hB, cB, hBbf, 0, H, nullptr, 0, 0, 0);
  }
  for (int s = 0; s < S; ++s) {  // L1 backward
    const int t = S - 1 - s;
    gemm(hBbf + H, DO, weU1b, H, gx_b + (size_t)t * G4H, (long long)S * G4H, nullptr,
         g_step, G4H, B, G4H);
    update(g_step, smask, S, t, hB, cB, hBbf, H, H, nullptr, 0, 0, 0);
  }

  // ---- decoder: hA/cA = finals of enc L0 (layer-1 state), hB/cB = enc L1 ----
  for (int t = 0; t < T; ++t) {
    // layer 1: g1 = gxd[:,t,:] + h1 @ dU0^T
    gemm(hAbf, DO, wdU0, DO, gxd + (size_t)t * G4D, (long long)T * G4D, nullptr,
         g_step, G4D, B, G4D);
    update(g_step, tmask, T, t, hA, cA, hAbf, 0, DO, nullptr, 0, 0, 0);
    // layer 2: g2 = db1 + h1 @ dW1^T, then += h2 @ dU1^T
    gemm(hAbf, DO, wdW1, DO, nullptr, 0, db1, g_step, G4D, B, G4D);
    gemm(hBbf, DO, wdU1, DO, g_step, G4D, nullptr, g_step, G4D, B, G4D);
    update(g_step, tmask, T, t, hB, cB, hBbf, 0, DO,
           dout_bf, (long long)T * DO, DO, 0);
  }

  // ---- output projection: logits = dout @ wW^T + wb  (dominant GEMM) ----
  gemm(dout_bf, DO, wwW, DO, nullptr, 0, wb, out, V, B * T, V);
}
